// DrivingController_52467320488378
// MI455X (gfx1250) — compile-verified, matrix operand order checked by reading, not run on hardware
//
#include <hip/hip_runtime.h>

// MI455X / gfx1250, wave32. Heavy GEMMs on v_wmma_f32_16x16x32_f16 (f32 acc),
// f16 activations, pre-swizzled B operands, TDM double-buffered B for pe1.

typedef _Float16 v8h  __attribute__((ext_vector_type(8)));
typedef _Float16 v16h __attribute__((ext_vector_type(16)));
typedef float    v8f  __attribute__((ext_vector_type(8)));
typedef unsigned int u32x4 __attribute__((ext_vector_type(4)));
typedef int          i32x4 __attribute__((ext_vector_type(4)));
typedef int          i32x8 __attribute__((ext_vector_type(8)));

#define DT_STEP 0.1f

#if __has_builtin(__builtin_amdgcn_tensor_load_to_lds)
#define HAVE_TDM 1
#else
#define HAVE_TDM 0
#endif

// ---------------------------------------------------------------------------
// B-matrix repack into per-lane WMMA order:
//   bsw[((kt*NT + nt)*32 + lane)*16 + j]  holds B[k][n] with
//   n = nt*16 + (lane&15), g0 = (lane>>4)*8,
//   k = kt*32 + (j<8 ? g0+j : g0+16+(j-8))
// ---------------------------------------------------------------------------

__global__ void repack_conv_w_kernel(const float* __restrict__ w,
                                     _Float16* __restrict__ bsw,
                                     int C_IN, int C_OUT) {
  const int K  = C_IN * 9;
  const int NT = C_OUT / 16;
  int tid = blockIdx.x * blockDim.x + threadIdx.x;
  if (tid >= K * C_OUT) return;
  int j    = tid & 15;
  int lane = (tid >> 4) & 31;
  int rest = tid >> 9;
  int nt = rest % NT;
  int kt = rest / NT;
  int n  = nt * 16 + (lane & 15);
  int g0 = (lane >> 4) * 8;
  int k  = kt * 32 + ((j < 8) ? (g0 + j) : (g0 + 16 + (j - 8)));
  int p = k / C_IN, c = k % C_IN;          // im2col order: (dy,dx) major, ci minor
  int dy = p / 3, dx = p % 3;
  bsw[tid] = (_Float16)w[((n * C_IN + c) * 3 + dy) * 3 + dx];  // OIHW source
}

__global__ void repack_pe1_w_kernel(const float* __restrict__ w,
                                    _Float16* __restrict__ bsw) {
  // K = 8192 in NHWC-flatten order (s*128 + c); reference rows are NCHW (c*64 + s)
  int tid = blockIdx.x * blockDim.x + threadIdx.x;
  if (tid >= 8192 * 256) return;
  int j    = tid & 15;
  int lane = (tid >> 4) & 31;
  int rest = tid >> 9;
  int nt = rest & 15;           // NT = 16
  int kt = rest >> 4;
  int n  = nt * 16 + (lane & 15);
  int g0 = (lane >> 4) * 8;
  int k  = kt * 32 + ((j < 8) ? (g0 + j) : (g0 + 16 + (j - 8)));
  int s = k >> 7;               // spatial y*8+x
  int c = k & 127;              // channel
  int kref = c * 64 + s;        // NCHW feature index
  bsw[tid] = (_Float16)w[kref * 256 + n];
}

__global__ void repack_pe2_w_kernel(const float* __restrict__ w,
                                    _Float16* __restrict__ bsw) {
  int tid = blockIdx.x * blockDim.x + threadIdx.x;
  if (tid >= 256 * 64) return;
  int j    = tid & 15;
  int lane = (tid >> 4) & 31;
  int rest = tid >> 9;
  int nt = rest & 3;            // NT = 4
  int kt = rest >> 2;
  int n  = nt * 16 + (lane & 15);
  int g0 = (lane >> 4) * 8;
  int k  = kt * 32 + ((j < 8) ? (g0 + j) : (g0 + 16 + (j - 8)));
  bsw[tid] = (_Float16)w[k * 64 + n];
}

// ---------------------------------------------------------------------------
// conv1: 3->32, s2 p1, fp32 VALU (K=27 too ragged for WMMA), out NHWC f16.
// ---------------------------------------------------------------------------
__global__ void conv1_kernel(const float* __restrict__ x,   // (512,3,64,64) f32
                             const float* __restrict__ w,   // (32,3,3,3)
                             const float* __restrict__ b,   // (32)
                             _Float16* __restrict__ out) {  // (512,32,32,32) NHWC f16
  __shared__ float ws[32 * 27];
  __shared__ float bs[32];
  for (int i = threadIdx.x; i < 32 * 27; i += blockDim.x) ws[i] = w[i];
  if (threadIdx.x < 32) bs[threadIdx.x] = b[threadIdx.x];
  __syncthreads();
  int pix = blockIdx.x * blockDim.x + threadIdx.x;  // 512*32*32
  if (pix >= 512 * 32 * 32) return;
  int img = pix >> 10;
  int oy  = (pix >> 5) & 31;
  int ox  = pix & 31;
  float acc[32];
#pragma unroll
  for (int co = 0; co < 32; ++co) acc[co] = bs[co];
  for (int dy = 0; dy < 3; ++dy) {
    int iy = oy * 2 + dy - 1;
    if (iy < 0 || iy >= 64) continue;
    for (int dx = 0; dx < 3; ++dx) {
      int ix = ox * 2 + dx - 1;
      if (ix < 0 || ix >= 64) continue;
      float i0 = x[((size_t)(img * 3 + 0) * 64 + iy) * 64 + ix];
      float i1 = x[((size_t)(img * 3 + 1) * 64 + iy) * 64 + ix];
      float i2 = x[((size_t)(img * 3 + 2) * 64 + iy) * 64 + ix];
      int p = dy * 3 + dx;
#pragma unroll
      for (int co = 0; co < 32; ++co) {
        const float* wp = &ws[co * 27];
        acc[co] = fmaf(i0, wp[p], fmaf(i1, wp[9 + p], fmaf(i2, wp[18 + p], acc[co])));
      }
    }
  }
  _Float16* o = out + (size_t)pix * 32;
#pragma unroll
  for (int co = 0; co < 32; ++co) o[co] = (_Float16)fmaxf(acc[co], 0.f);
}

// ---------------------------------------------------------------------------
// Implicit-GEMM stride-2 3x3 conv via WMMA, NHWC f16.
// One wave computes 16(M) x (16*NTPW)(N); A reused across NTPW WMMAs.
// Padding via clamped-address load + select-to-zero (keeps EXEC full).
// ---------------------------------------------------------------------------
template <int C_IN, int H_IN, int W_IN>
__device__ __forceinline__ v8h conv_patch8(const _Float16* __restrict__ in,
                                           int img, int oy, int ox, int k) {
  int p = k / C_IN, c = k % C_IN;    // group of 8 stays inside one (dy,dx) run
  int dy = p / 3, dx = p % 3;
  int iy = oy * 2 + dy - 1;
  int ix = ox * 2 + dx - 1;
  bool valid = (iy >= 0) & (iy < H_IN) & (ix >= 0) & (ix < W_IN);
  int iyc = iy < 0 ? 0 : (iy >= H_IN ? H_IN - 1 : iy);
  int ixc = ix < 0 ? 0 : (ix >= W_IN ? W_IN - 1 : ix);
  v8h v = *(const v8h*)(in + ((size_t)(img * H_IN + iyc) * W_IN + ixc) * C_IN + c);
  v8h z = {};
  return valid ? v : z;
}

template <int C_IN, int C_OUT, int H_IN, int W_IN, int NTPW>
__global__ void conv_s2_wmma_kernel(const _Float16* __restrict__ in,
                                    const _Float16* __restrict__ bsw,
                                    const float* __restrict__ bias,
                                    _Float16* __restrict__ out) {
  constexpr int HO = H_IN / 2, WO = W_IN / 2;
  constexpr int KT = (C_IN * 9) / 32;
  constexpr int NT = C_OUT / 16;
  constexpr int NG = NT / NTPW;
  constexpr int M  = 512 * HO * WO;
  constexpr int TILES = (M / 16) * NG;
  const int lane = threadIdx.x & 31;
  const int tile = blockIdx.x * (blockDim.x >> 5) + (threadIdx.x >> 5);
  if (tile >= TILES) return;
  const int mtile = tile / NG;
  const int nt0   = (tile % NG) * NTPW;
  const int ma  = mtile * 16 + (lane & 15);       // A-row (pixel) this lane feeds
  const int img = ma / (HO * WO);
  const int rem = ma % (HO * WO);
  const int oy  = rem / WO, ox = rem % WO;
  const int g0  = (lane >> 4) * 8;                // K-group split across lane halves
  v8f acc[NTPW];
#pragma unroll
  for (int q = 0; q < NTPW; ++q) acc[q] = (v8f){};
  for (int kt = 0; kt < KT; ++kt) {
    v8h a0 = conv_patch8<C_IN, H_IN, W_IN>(in, img, oy, ox, kt * 32 + g0);
    v8h a1 = conv_patch8<C_IN, H_IN, W_IN>(in, img, oy, ox, kt * 32 + g0 + 16);
    v16h a;
#pragma unroll
    for (int i = 0; i < 8; ++i) { a[i] = a0[i]; a[8 + i] = a1[i]; }
#pragma unroll
    for (int q = 0; q < NTPW; ++q) {
      v16h bm = *(const v16h*)(bsw + (((size_t)kt * NT + nt0 + q) * 32 + lane) * 16);
      acc[q] = __builtin_amdgcn_wmma_f32_16x16x32_f16(false, a, false, bm,
                                                      (short)0, acc[q], false, false);
    }
    if (kt + 1 < KT)
      __builtin_prefetch(bsw + (((size_t)(kt + 1) * NT + nt0) * 32 + lane) * 16, 0, 1);
  }
  const int mbase = mtile * 16 + ((lane >> 4) << 3);  // D rows: M = r + 8*(lane>>4)
#pragma unroll
  for (int q = 0; q < NTPW; ++q) {
    const int n = (nt0 + q) * 16 + (lane & 15);
    const float bn = bias[n];
#pragma unroll
    for (int r = 0; r < 8; ++r) {
      float v = fmaxf(acc[q][r] + bn, 0.f);
      out[(size_t)(mbase + r) * C_OUT + n] = (_Float16)v;
    }
  }
}

// ---------------------------------------------------------------------------
// Generic dense GEMM via WMMA with N-tile blocking (used for pe2).
// ---------------------------------------------------------------------------
template <bool RELU, bool OUTF16, int NTPW>
__global__ void gemm_wmma_kernel(const _Float16* __restrict__ A,
                                 const _Float16* __restrict__ bsw,
                                 const float* __restrict__ bias,
                                 void* __restrict__ outp,
                                 int Mtiles, int NT, int KT) {
  const int lane = threadIdx.x & 31;
  const int tile = blockIdx.x * (blockDim.x >> 5) + (threadIdx.x >> 5);
  const int NG = NT / NTPW;
  if (tile >= Mtiles * NG) return;
  const int mtile = tile / NG;
  const int nt0   = (tile % NG) * NTPW;
  const int K = KT * 32, N = NT * 16;
  const _Float16* arow = A + (size_t)(mtile * 16 + (lane & 15)) * K + ((lane >> 4) << 3);
  v8f acc[NTPW];
#pragma unroll
  for (int q = 0; q < NTPW; ++q) acc[q] = (v8f){};
  for (int kt = 0; kt < KT; ++kt) {
    v8h a0 = *(const v8h*)(arow + kt * 32);
    v8h a1 = *(const v8h*)(arow + kt * 32 + 16);
    v16h a;
#pragma unroll
    for (int i = 0; i < 8; ++i) { a[i] = a0[i]; a[8 + i] = a1[i]; }
#pragma unroll
    for (int q = 0; q < NTPW; ++q) {
      v16h bm = *(const v16h*)(bsw + (((size_t)kt * NT + nt0 + q) * 32 + lane) * 16);
      acc[q] = __builtin_amdgcn_wmma_f32_16x16x32_f16(false, a, false, bm,
                                                      (short)0, acc[q], false, false);
    }
    if (kt + 1 < KT) __builtin_prefetch(arow + (size_t)(kt + 1) * 32, 0, 1);
  }
  const int mbase = mtile * 16 + ((lane >> 4) << 3);
#pragma unroll
  for (int q = 0; q < NTPW; ++q) {
    const int n = (nt0 + q) * 16 + (lane & 15);
    const float bn = bias[n];
#pragma unroll
    for (int r = 0; r < 8; ++r) {
      float v = acc[q][r] + bn;
      if (RELU) v = fmaxf(v, 0.f);
      size_t oi = (size_t)(mbase + r) * N + n;
      if (OUTF16) ((_Float16*)outp)[oi] = (_Float16)v;
      else        ((float*)outp)[oi] = v;
    }
  }
}

// ---------------------------------------------------------------------------
// pe1 GEMM (M=512, N=256, K=8192): TDM double-buffers the swizzled B panel
// through LDS (tensor_load_to_lds + s_wait_tensorcnt); 4 waves/block, each
// wave owns one N-tile; A from global, B from LDS (ds_load_b128).
// Chunk = 4 K-tiles x 4 N-tiles = 16 KB; 2 buffers = 32 KB LDS.
// ---------------------------------------------------------------------------
#define PE1_KT   256
#define PE1_NT   16
#define CHUNK_KT 4
#define NCHUNK   (PE1_KT / CHUNK_KT)
#define CHUNK_HALFS (CHUNK_KT * 4 * 512)   // 8192 halfs = 16 KB

#if HAVE_TDM
__device__ __forceinline__ void pe1_tdm_issue(const _Float16* gptr, unsigned lds_off) {
  unsigned long long ga = (unsigned long long)(uintptr_t)gptr;
  u32x4 g0;
  g0[0] = 1u;                                   // count=1, user mode
  g0[1] = lds_off;                              // lds_addr (bytes)
  g0[2] = (unsigned)(ga & 0xFFFFFFFFu);         // global_addr[31:0]
  g0[3] = (unsigned)((ga >> 32) & 0x1FFFFFFu) | (2u << 30);  // addr[56:32] | type=2
  i32x8 g1;
  g1[0] = 3 << 16;                              // data_size = 8 bytes
  g1[1] = 512 << 16;                            // tensor_dim0[15:0]  (=512 units)
  g1[2] = CHUNK_KT << 16;                       // tensor_dim0 hi | tensor_dim1 lo
  g1[3] = 512 << 16;                            // tensor_dim1 hi | tile_dim0 = 512
  g1[4] = CHUNK_KT;                             // tile_dim1 | tile_dim2=0
  g1[5] = 2048;                                 // tensor_dim0_stride[31:0] (16KB/8B)
  g1[6] = 0;
  g1[7] = 0;
  i32x4 gz = {0, 0, 0, 0};
#if defined(__clang_major__) && __clang_major__ >= 23
  i32x8 gz8 = {0, 0, 0, 0, 0, 0, 0, 0};
  __builtin_amdgcn_tensor_load_to_lds(g0, g1, gz, gz, gz8, 0);
#else
  __builtin_amdgcn_tensor_load_to_lds(g0, g1, gz, gz, 0);
#endif
}
#endif

__global__ void gemm_pe1_tdm_kernel(const _Float16* __restrict__ A,    // (512,8192) f16
                                    const _Float16* __restrict__ bsw,  // swizzled B
                                    const float* __restrict__ bias,
                                    _Float16* __restrict__ out) {      // (512,256) f16
  __shared__ _Float16 bsh[2][CHUNK_HALFS];
  const int lane = threadIdx.x & 31;
  const int wv   = threadIdx.x >> 5;           // 0..3, owns N-tile nt0+wv
  const int mtile = blockIdx.x >> 2;           // 32 M-tiles
  const int nt0   = (blockIdx.x & 3) * 4;      // 4 N-groups of 4 tiles
  const _Float16* arow =
      A + (size_t)(mtile * 16 + (lane & 15)) * 8192 + ((lane >> 4) << 3);
  // global byte address of chunk c for this block's N-group:
  //   bsw + (kt0*16 + nt0) * 1024 bytes, kt0 = c*CHUNK_KT
  const _Float16* gbase = bsw + (size_t)nt0 * 512;

#if HAVE_TDM
  if (threadIdx.x < 32) {
    pe1_tdm_issue(gbase, (unsigned)(uintptr_t)&bsh[0][0]);
    __builtin_amdgcn_s_wait_tensorcnt(0);
  }
  __syncthreads();
#else
  for (int i = threadIdx.x; i < CHUNK_HALFS / 8; i += blockDim.x) {
    int kt = i / 256, rj = i % 256;            // row kt, 256 x v8h per row
    ((v8h*)&bsh[0][0])[i] =
        *(const v8h*)(gbase + ((size_t)kt * 16) * 512 + rj * 8);
  }
  __syncthreads();
#endif

  v8f acc = {};
  for (int c = 0; c < NCHUNK; ++c) {
    const int buf = c & 1;
#if HAVE_TDM
    if (c + 1 < NCHUNK && threadIdx.x < 32)
      pe1_tdm_issue(gbase + (size_t)(c + 1) * CHUNK_KT * 16 * 512,
                    (unsigned)(uintptr_t)&bsh[1 - buf][0]);
#else
    if (c + 1 < NCHUNK) {
      for (int i = threadIdx.x; i < CHUNK_HALFS / 8; i += blockDim.x) {
        int kt = i / 256, rj = i % 256;
        ((v8h*)&bsh[1 - buf][0])[i] =
            *(const v8h*)(gbase + ((size_t)((c + 1) * CHUNK_KT + kt) * 16) * 512 +
                          rj * 8);
      }
    }
#endif
#pragma unroll
    for (int ktl = 0; ktl < CHUNK_KT; ++ktl) {
      const int kt = c * CHUNK_KT + ktl;
      v8h a0 = *(const v8h*)(arow + kt * 32);
      v8h a1 = *(const v8h*)(arow + kt * 32 + 16);
      v16h a;
#pragma unroll
      for (int i = 0; i < 8; ++i) { a[i] = a0[i]; a[8 + i] = a1[i]; }
      const _Float16* bp = &bsh[buf][(ktl * 4 + wv) * 512 + lane * 16];
      v16h bm = *(const v16h*)bp;
      acc = __builtin_amdgcn_wmma_f32_16x16x32_f16(false, a, false, bm,
                                                   (short)0, acc, false, false);
    }
#if HAVE_TDM
    if (threadIdx.x < 32) __builtin_amdgcn_s_wait_tensorcnt(0);
#endif
    __syncthreads();
  }
  const int n  = (nt0 + wv) * 16 + (lane & 15);
  const float bn = bias[n];
  const int mbase = mtile * 16 + ((lane >> 4) << 3);
#pragma unroll
  for (int r = 0; r < 8; ++r) {
    float v = fmaxf(acc[r] + bn, 0.f);       // pe1 has ReLU
    out[(size_t)(mbase + r) * 256 + n] = (_Float16)v;
  }
}

// ---------------------------------------------------------------------------
// LGTCN scan + decoder. N=1 collapses the graph conv to h @ (th0+th1+th2).
// ---------------------------------------------------------------------------
__global__ void rnn_kernel(const float* __restrict__ u,        // (8*64, 64) rows = b*64+t
                           const float* __restrict__ theta,    // (3,64,64)
                           const float* __restrict__ ufw,      // (64,64)
                           const float* __restrict__ bfv,      // (64)
                           const float* __restrict__ tau_raw,  // (64)
                           const float* __restrict__ abias,    // (64)
                           const float* __restrict__ d1w, const float* __restrict__ d1b,
                           const float* __restrict__ d2w, const float* __restrict__ d2b,
                           const float* __restrict__ d3w, const float* __restrict__ d3b,
                           float* __restrict__ out) {          // 1024 controls + 512 hT
  __shared__ float Th[64 * 64];
  __shared__ float Uf[64 * 64];
  __shared__ float sb[64], sa[64], sit[64];
  __shared__ float h[8 * 64];
  __shared__ float f[8 * 64];
  __shared__ float c1[8 * 128];
  __shared__ float c2[8 * 64];
  const int tid = threadIdx.x;  // 512
  for (int i = tid; i < 4096; i += 512) {
    Th[i] = theta[i] + theta[4096 + i] + theta[8192 + i];
    Uf[i] = ufw[i];
  }
  if (tid < 64) {
    sb[tid]  = bfv[tid];
    sa[tid]  = abias[tid];
    float tau = log1pf(expf(tau_raw[tid]));  // softplus
    sit[tid] = 1.0f / tau;
  }
  if (tid < 512) h[tid] = 0.f;
  __syncthreads();
  const int b = tid >> 6, d = tid & 63;
  for (int t = 0; t < 64; ++t) {
    {
      float acc = sb[d];
      const float* urow = u + ((size_t)(b * 64 + t)) * 64;
      for (int m = 0; m < 64; ++m)
        acc = fmaf(h[b * 64 + m], Th[m * 64 + d], fmaf(urow[m], Uf[m * 64 + d], acc));
      f[tid] = 1.0f / (1.0f + expf(-acc));
    }
    __syncthreads();
    {
      float fv = f[tid];
      h[tid] = (h[tid] + DT_STEP * fv * sa[d]) / (1.0f + DT_STEP * (sit[d] + fv));
    }
    __syncthreads();
    for (int i = tid; i < 1024; i += 512) {
      int bb = i >> 7, j = i & 127;
      float acc = d1b[j];
      for (int m = 0; m < 64; ++m) acc = fmaf(h[bb * 64 + m], d1w[m * 128 + j], acc);
      c1[i] = fmaxf(acc, 0.f);
    }
    __syncthreads();
    {
      float acc = d2b[d];
      for (int m = 0; m < 128; ++m) acc = fmaf(c1[b * 128 + m], d2w[m * 64 + d], acc);
      c2[tid] = fmaxf(acc, 0.f);
    }
    __syncthreads();
    if (tid < 16) {
      int bb = tid >> 1, j = tid & 1;
      float acc = d3b[j];
      for (int m = 0; m < 64; ++m) acc = fmaf(c2[bb * 64 + m], d3w[m * 2 + j], acc);
      out[(size_t)(bb * 64 + t) * 2 + j] = acc;
    }
    __syncthreads();
  }
  if (tid < 512) out[1024 + tid] = h[tid];
}

// ---------------------------------------------------------------------------
extern "C" void kernel_launch(void* const* d_in, const int* in_sizes, int n_in,
                              void* d_out, int out_size, void* d_ws, size_t ws_size,
                              hipStream_t stream) {
  (void)in_sizes; (void)n_in; (void)out_size; (void)ws_size;
  const float* frames = (const float*)d_in[0];
  const float* c1w = (const float*)d_in[1];
  const float* c1b = (const float*)d_in[2];
  const float* c2w = (const float*)d_in[3];
  const float* c2b = (const float*)d_in[4];
  const float* c3w = (const float*)d_in[5];
  const float* c3b = (const float*)d_in[6];
  const float* p1w = (const float*)d_in[7];
  const float* p1b = (const float*)d_in[8];
  const float* p2w = (const float*)d_in[9];
  const float* p2b = (const float*)d_in[10];
  const float* th  = (const float*)d_in[11];
  const float* ufw = (const float*)d_in[12];
  const float* bfv = (const float*)d_in[13];
  const float* tau = (const float*)d_in[14];
  const float* ab  = (const float*)d_in[15];
  const float* d1w = (const float*)d_in[16];
  const float* d1b = (const float*)d_in[17];
  const float* d2w = (const float*)d_in[18];
  const float* d2b = (const float*)d_in[19];
  const float* d3w = (const float*)d_in[20];
  const float* d3b = (const float*)d_in[21];

  char* ws = (char*)d_ws;
  size_t off = 0;
  auto walloc = [&](size_t bytes) -> void* {
    void* p = ws + off;
    off = (off + bytes + 255) & ~(size_t)255;
    return p;
  };
  _Float16* act1  = (_Float16*)walloc(512ull * 32 * 32 * 32 * 2);   // NHWC f16
  _Float16* act2  = (_Float16*)walloc(512ull * 16 * 16 * 64 * 2);
  _Float16* act3  = (_Float16*)walloc(512ull * 8 * 8 * 128 * 2);
  _Float16* pe1o  = (_Float16*)walloc(512ull * 256 * 2);
  float*    uo    = (float*)walloc(512ull * 64 * 4);
  _Float16* w2sw  = (_Float16*)walloc(288ull * 64 * 2);
  _Float16* w3sw  = (_Float16*)walloc(576ull * 128 * 2);
  _Float16* p1sw  = (_Float16*)walloc(8192ull * 256 * 2);
  _Float16* p2sw  = (_Float16*)walloc(256ull * 64 * 2);

  // weight repacks (B swizzle + f16 cast; pe1 also folds NCHW->NHWC perm)
  repack_conv_w_kernel<<<(288 * 64 + 255) / 256, 256, 0, stream>>>(c2w, w2sw, 32, 64);
  repack_conv_w_kernel<<<(576 * 128 + 255) / 256, 256, 0, stream>>>(c3w, w3sw, 64, 128);
  repack_pe1_w_kernel<<<(8192 * 256) / 256, 256, 0, stream>>>(p1w, p1sw);
  repack_pe2_w_kernel<<<(256 * 64) / 256, 256, 0, stream>>>(p2w, p2sw);

  // conv1 (fp32 VALU), then WMMA convs + MLPs
  conv1_kernel<<<(512 * 32 * 32) / 128, 128, 0, stream>>>(frames, c1w, c1b, act1);
  // conv2: 8192 M-tiles x 1 N-group (NTPW=4 covers all 64 ch)
  conv_s2_wmma_kernel<32, 64, 32, 32, 4><<<8192 / 4, 128, 0, stream>>>(
      act1, w2sw, c2b, act2);
  // conv3: 2048 M-tiles x 2 N-groups
  conv_s2_wmma_kernel<64, 128, 16, 16, 4><<<4096 / 4, 128, 0, stream>>>(
      act2, w3sw, c3b, act3);
  // pe1 via TDM double-buffered LDS: 32 M-tiles x 4 N-groups, 128 thr/blk
  gemm_pe1_tdm_kernel<<<128, 128, 0, stream>>>(act3, p1sw, p1b, pe1o);
  // pe2: 32 M-tiles x 1 N-group
  gemm_wmma_kernel<false, false, 4><<<8, 128, 0, stream>>>(
      pe1o, p2sw, p2b, (void*)uo, 32, 4, 8);

  // sequential liquid-RNN + decoder, single workgroup
  rnn_kernel<<<1, 512, 0, stream>>>(uo, th, ufw, bfv, tau, ab,
                                    d1w, d1b, d2w, d2b, d3w, d3b, (float*)d_out);
}
